// MultiHeadAttention_61667140436614
// MI455X (gfx1250) — compile-verified, matrix operand order checked by reading, not run on hardware
//
#include <hip/hip_runtime.h>
#include <hip/hip_bf16.h>

typedef __attribute__((ext_vector_type(16))) _Float16 v16h;
typedef __attribute__((ext_vector_type(4)))  _Float16 h4;
typedef __attribute__((ext_vector_type(8)))  float    v8f;
typedef __attribute__((ext_vector_type(4)))  unsigned int v4u;
typedef __attribute__((ext_vector_type(8)))  int      v8i;
typedef __attribute__((ext_vector_type(4)))  int      v4i;

#define NB 8192
#define SQ 12
#define NH 8
#define DK 64
#define DM 512    // NH*DK
#define SSTR 72   // padded row stride of the neighbor-sum buffer (halo 4 front / 4 back)

// f16 A/B operand K-index pattern for v_wmma_*_16x16x32_f16 (wave32):
//   element e (0..15) of v16h -> VGPR j=e/2, half off=e%2
//   lanes 0-15: K = {0..7, 16..23};  lanes 16-31: +8
__device__ __forceinline__ int kmap(int e, int hi) {
    int j = e >> 1, off = e & 1;
    int base = (j < 4) ? (2 * j) : (16 + 2 * (j - 4));
    return base + off + 8 * hi;
}

__device__ __forceinline__ v8f wmma_f16(const v16h& a, const v16h& b, const v8f& c) {
    return __builtin_amdgcn_wmma_f32_16x16x32_f16(false, a, false, b, (short)0, c, false, false);
}

__global__ __launch_bounds__(256)
void mha_fused_kernel(const float* __restrict__ query,
                      const float* __restrict__ key,
                      const float* __restrict__ value,
                      const int*   __restrict__ mask,
                      const float* __restrict__ conv_w,   // [12][12][3] -> row m: 36 contiguous
                      const float* __restrict__ conv_b,   // [12]
                      const float* __restrict__ W_out,    // [512][512], out = x @ W^T
                      const float* __restrict__ b_out,    // [512]
                      float*       __restrict__ out)      // [B][12][512]
{
    // conv outputs per head/tensor; first 16 KB later overlaid by x[16][512] f16
    __shared__ __align__(16) _Float16 conv_lds[NH][3][16][DK];   // 48 KB
    // per-head scratch: neighbor-sum s_pad[12][72] (halo-padded); later overlaid by
    // scores float[12][16] (bytes 0..767) and p f16[12][12] (half idx 384..527)
    __shared__ __align__(16) _Float16 scratch[NH][SQ * SSTR];    // 13.5 KB
    __shared__ __align__(16) int mask_lds[SQ * SQ];              // 576 B (TDM destination)

    const int tid  = threadIdx.x;
    const int lane = tid & 31;
    const int w    = tid >> 5;          // wave id == head id
    const int h    = w;
    const int hi   = lane >> 4;
    const int nl   = lane & 15;
    const int b    = blockIdx.x;

    const float* qkv[3] = { query + (size_t)b * SQ * DM,
                            key   + (size_t)b * SQ * DM,
                            value + (size_t)b * SQ * DM };

    // ---- TDM: DMA the 12x12 int mask tile for this batch into LDS (wave 0) ----
    if (w == 0) {
        unsigned long long ga = (unsigned long long)(uintptr_t)(mask + (size_t)b * (SQ * SQ));
        unsigned lds_off = (unsigned)(uintptr_t)(void*)mask_lds;
        v4u g0;
        g0[0] = 1u;                                     // count=1, user descriptor
        g0[1] = lds_off;                                // lds_addr
        g0[2] = (unsigned)(ga & 0xffffffffu);           // global_addr[31:0]
        g0[3] = (unsigned)((ga >> 32) & 0x01ffffffu) | (2u << 30);  // addr[56:32] | type=2
        v8i g1;
        g1[0] = (int)(2u << 16);        // workgroup_mask=0, data_size=2 (4 bytes)
        g1[1] = (int)(144u << 16);      // tensor_dim0 = 144 (low 16 at bits 63:48)
        g1[2] = (int)(1u << 16);        // tensor_dim0 hi=0 | tensor_dim1 = 1
        g1[3] = (int)(144u << 16);      // tensor_dim1 hi=0 | tile_dim0 = 144
        g1[4] = 1;                      // tile_dim1 = 1, tile_dim2 = 0
        g1[5] = 144;                    // tensor_dim0_stride low 32
        g1[6] = 0;
        g1[7] = 0;
        v4i gz = {0, 0, 0, 0};
#if __clang_major__ >= 23
        v8i gz8 = {0, 0, 0, 0, 0, 0, 0, 0};
        __builtin_amdgcn_tensor_load_to_lds(g0, g1, gz, gz, gz8, 0);
#else
        __builtin_amdgcn_tensor_load_to_lds(g0, g1, gz, gz, 0);
#endif
    }

    // Prefetch the projection weights toward L2 (global_prefetch_b8 path).
    #pragma unroll
    for (int i = 0; i < 4; ++i)
        __builtin_prefetch(&W_out[((size_t)tid * 4 + i) * 256], 0, 1);

    // Per-lane conv bias for C-layout rows m = r + 8*hi
    float cbias[8];
    #pragma unroll
    for (int r = 0; r < 8; ++r) {
        int m = r + 8 * hi;
        cbias[r] = (m < SQ) ? conv_b[m] : 0.0f;
    }

    // Conv im2col A-operand: A(m, k) = conv_w[m*36 + k], k = s_in*3 + t < 36, m < 12
    v16h a_conv[2];
    #pragma unroll
    for (int ks = 0; ks < 2; ++ks) {
        #pragma unroll
        for (int e = 0; e < 16; ++e) {
            int k = ks * 32 + kmap(e, hi);
            a_conv[ks][e] = (nl < SQ && k < 36) ? (_Float16)conv_w[nl * 36 + k] : (_Float16)0;
        }
    }

    _Float16* s_pad = &scratch[h][0];   // [12][SSTR], data at column offset +4

    // Zero the halo columns once (pos 0..3 and 68..71 of each row)
    #pragma unroll
    for (int j = lane; j < 96; j += 32) {
        int row = j >> 3, c = j & 7;
        int pos = (c < 4) ? c : (64 + c);
        s_pad[row * SSTR + pos] = (_Float16)0;
    }

    // ---- Stage 1: neighbor-sum + Conv1d (as GEMM) for q, k, v ----
    for (int t = 0; t < 3; ++t) {
        const float* src = qkv[t];
        // s[s_in][d] = sum over heads {h-1,h,h+1}; branchless: clamped row * {0,1} weight
        #pragma unroll
        for (int i = 0; i < 6; ++i) {
            int idx = lane + 32 * i;            // 192 groups of 4 floats (12*64 total)
            int sin = idx >> 4;
            int d4  = (idx & 15) * 4;
            float ax = 0.f, ay = 0.f, az = 0.f, aw = 0.f;
            #pragma unroll
            for (int dh = -1; dh <= 1; ++dh) {
                int   hh  = h + dh;
                float wgt = (hh >= 0 && hh < NH) ? 1.0f : 0.0f;
                int   hc  = (hh < 0) ? 0 : ((hh > NH - 1) ? (NH - 1) : hh);
                const float4 vv = *(const float4*)(src + sin * DM + hc * DK + d4);
                ax += wgt * vv.x; ay += wgt * vv.y; az += wgt * vv.z; aw += wgt * vv.w;
            }
            h4 hv = { (_Float16)ax, (_Float16)ay, (_Float16)az, (_Float16)aw };
            *(h4*)&s_pad[sin * SSTR + 4 + d4] = hv;
        }
        __syncthreads();

        // conv GEMM: out[16x64] = A[16xK] x patch[Kx64], 4 N-tiles x 2 K-steps
        #pragma unroll
        for (int nt = 0; nt < 4; ++nt) {
            v8f c = {};
            #pragma unroll
            for (int ks = 0; ks < 2; ++ks) {
                v16h bf;
                #pragma unroll
                for (int e = 0; e < 16; ++e) {
                    int k  = ks * 32 + kmap(e, hi);
                    int kc = (k < 36) ? k : 35;        // clamp -> unconditional DS load
                    int sin = kc / 3, tt = kc - 3 * sin;
                    _Float16 v = s_pad[sin * SSTR + 4 + nt * 16 + nl + tt - 1];
                    bf[e] = (k < 36) ? v : (_Float16)0;
                }
                c = wmma_f16(a_conv[ks], bf, c);
            }
            #pragma unroll
            for (int r = 0; r < 8; ++r) {
                int m = r + 8 * hi;               // rows >=12 hold exact zeros
                conv_lds[h][t][m][nt * 16 + nl] = (_Float16)(c[r] + cbias[r]);
            }
        }
        __syncthreads();
    }

    // ---- Stage 2: scores = q . k^T  (K = 64, two steps) ----
    v8f sc = {};
    #pragma unroll
    for (int ks = 0; ks < 2; ++ks) {
        v16h aq, bk;
        #pragma unroll
        for (int e = 0; e < 16; ++e) {
            int k = ks * 32 + kmap(e, hi);
            aq[e] = conv_lds[h][0][nl][k];    // A(m,k) = q[m][k]
            bk[e] = conv_lds[h][1][nl][k];    // B(k,n) = Kmat[n][k]
        }
        sc = wmma_f16(aq, bk, sc);
    }
    __syncthreads();

    float* sco = (float*)&scratch[h][0];          // overlay: scores[12][16]
    _Float16* p_ptr = &scratch[h][384];           // overlay: p[12][12] f16
    #pragma unroll
    for (int r = 0; r < 8; ++r) {
        int m = r + 8 * hi;
        if (m < SQ) sco[m * 16 + nl] = sc[r];
    }
    __builtin_amdgcn_s_wait_tensorcnt(0);          // mask tile landed in LDS
    __syncthreads();

    // ---- Stage 3: masked softmax over 12 keys (one lane per query row) ----
    if (lane < SQ) {
        float vals[SQ];
        float mx = -3.4e38f;
        #pragma unroll
        for (int j = 0; j < SQ; ++j) {
            float v = sco[lane * 16 + j] * 0.125f;      // 1/sqrt(64)
            if (mask_lds[lane * SQ + j] == 0) v = -1e9f;
            vals[j] = v;
            mx = fmaxf(mx, v);
        }
        float sum = 0.0f;
        #pragma unroll
        for (int j = 0; j < SQ; ++j) { vals[j] = __expf(vals[j] - mx); sum += vals[j]; }
        float inv = 1.0f / sum;
        #pragma unroll
        for (int j = 0; j < SQ; ++j) p_ptr[lane * SQ + j] = (_Float16)(vals[j] * inv);
    }
    __syncthreads();

    // ---- Stage 4: ctx = p . v  (K = 12, padded into one 32-step) ----
    v16h ap;
    #pragma unroll
    for (int e = 0; e < 16; ++e) {
        int k  = kmap(e, hi);
        int kc = (k < SQ) ? k : (SQ - 1);
        int mc = (nl < SQ) ? nl : (SQ - 1);
        _Float16 v = p_ptr[mc * SQ + kc];
        ap[e] = (nl < SQ && k < SQ) ? v : (_Float16)0;
    }
    v8f ctx[4];
    #pragma unroll
    for (int nt = 0; nt < 4; ++nt) {
        v16h bv;
        #pragma unroll
        for (int e = 0; e < 16; ++e) {
            int k  = kmap(e, hi);
            int kc = (k < SQ) ? k : (SQ - 1);
            _Float16 v = conv_lds[h][2][kc][nt * 16 + nl];
            bv[e] = (k < SQ) ? v : (_Float16)0;
        }
        v8f z = {};
        ctx[nt] = wmma_f16(ap, bv, z);
    }
    __syncthreads();   // all waves finished reading conv_lds; safe to overlay x

    // ---- Stage 5: store context as f16 x[16][512] (rows 12-15 are zeros) ----
    _Float16 (*x_lds)[DM] = (_Float16 (*)[DM]) &conv_lds[0][0][0][0];   // 16 KB overlay
    #pragma unroll
    for (int nt = 0; nt < 4; ++nt) {
        #pragma unroll
        for (int r = 0; r < 8; ++r) {
            int m = r + 8 * hi;
            x_lds[m][h * DK + nt * 16 + nl] = (_Float16)ctx[nt][r];
        }
    }
    __syncthreads();

    // ---- Stage 6: projection out[12x512] = x @ W^T + b (wave w owns 4 N-tiles) ----
    v8f acc[4] = {{}, {}, {}, {}};
    #pragma unroll 4
    for (int K0 = 0; K0 < DM; K0 += 32) {
        v16h ax;
        #pragma unroll
        for (int e = 0; e < 16; ++e) ax[e] = x_lds[nl][K0 + kmap(e, hi)];
        #pragma unroll
        for (int i = 0; i < 4; ++i) {
            int ng = (w * 4 + i) * 16 + nl;            // output column
            v16h bw;
            #pragma unroll
            for (int e = 0; e < 16; ++e)               // B(k,n) = W_out[n][k]; 8-wide runs
                bw[e] = (_Float16)W_out[(size_t)ng * DM + K0 + kmap(e, hi)];
            acc[i] = wmma_f16(ax, bw, acc[i]);
        }
    }
    #pragma unroll
    for (int i = 0; i < 4; ++i) {
        int ng = (w * 4 + i) * 16 + nl;
        float bo = b_out[ng];
        #pragma unroll
        for (int r = 0; r < 8; ++r) {
            int m = r + 8 * hi;
            if (m < SQ)
                out[((size_t)b * SQ + m) * DM + ng] = acc[i][r] + bo;
        }
    }
}

extern "C" void kernel_launch(void* const* d_in, const int* in_sizes, int n_in,
                              void* d_out, int out_size, void* d_ws, size_t ws_size,
                              hipStream_t stream) {
    const float* query  = (const float*)d_in[0];
    const float* key    = (const float*)d_in[1];
    const float* value  = (const float*)d_in[2];
    const int*   mask   = (const int*)  d_in[3];
    const float* conv_w = (const float*)d_in[4];
    const float* conv_b = (const float*)d_in[5];
    const float* W_out  = (const float*)d_in[6];
    const float* b_out  = (const float*)d_in[7];
    float* out = (float*)d_out;
    (void)in_sizes; (void)n_in; (void)out_size; (void)d_ws; (void)ws_size;

    mha_fused_kernel<<<NB, 256, 0, stream>>>(query, key, value, mask,
                                             conv_w, conv_b, W_out, b_out, out);
}
